// Attention_18674517803116
// MI455X (gfx1250) — matrix operand order repaired and checked by reading, not compiled, not run on hardware
//
#include <hip/hip_runtime.h>
#include <hip/hip_bf16.h>
#include <math.h>

typedef _Float16 half_t;
typedef __attribute__((ext_vector_type(16))) _Float16 v16h;
typedef __attribute__((ext_vector_type(8)))  float    v8f;

#define B_  2
#define S_  2048
#define D_  512
#define H_  8
#define DH_ 64
#define N3_ 1536

__device__ __forceinline__ v8f wmma16(v16h a, v16h b, v8f c) {
  return __builtin_amdgcn_wmma_f32_16x16x32_f16(false, a, false, b, (short)0, c,
                                                false, false);
}

// A fragment: 16(M) x 32(K) f16.  lane<16: row=lane, K = i<8 ? i : i+8
//                                 lane>=16: row=lane-16, K offset +8
template<int STRIDE>
__device__ __forceinline__ v16h load_a_frag(const half_t* base, int lane) {
  const int m  = lane & 15;
  const int kb = (lane >> 4) * 8;
  const half_t* p = base + m * STRIDE + kb;
  v16h a;
#pragma unroll
  for (int i = 0; i < 8; ++i)  a[i] = p[i];
#pragma unroll
  for (int i = 8; i < 16; ++i) a[i] = p[i + 8];   // K = kb + 16 .. kb + 23
  return a;
}

// B fragment: 32(K) x 16(N) f16, logical B[K][N] stored as Mem[N][K]
// lanes 0-15: N=lane, K=0..7,16..23 ; lanes 16-31: N=lane-16, K=8..15,24..31
template<int STRIDE>
__device__ __forceinline__ v16h load_b_frag_nk(const half_t* base, int lane) {
  const int n  = lane & 15;
  const int kb = (lane >> 4) * 8;
  const half_t* p = base + n * STRIDE + kb;
  v16h b;
#pragma unroll
  for (int i = 0; i < 16; ++i) b[i] = p[i < 8 ? i : i + 8];
  return b;
}

// B fragment with logical B[K][N] stored as Mem[K][N]
template<int STRIDE>
__device__ __forceinline__ v16h load_b_frag_kn(const half_t* base, int lane) {
  const int n  = lane & 15;
  const int kb = (lane >> 4) * 8;
  v16h b;
#pragma unroll
  for (int i = 0; i < 16; ++i) b[i] = base[(kb + (i < 8 ? i : i + 8)) * STRIDE + n];
  return b;
}

// ---------------------------------------------------------------------------
// Kernel 1: qkv = x @ W^T + b ; scatter to head-major f16 q/k/v; q *= 1/8.
// grid = (1536/128, 4096/64), block = 256 (8 waves: 4 M-tiles x 2 N-strips)
// ---------------------------------------------------------------------------
__global__ __launch_bounds__(256)
void qkv_gemm_kernel(const float* __restrict__ x, const float* __restrict__ W,
                     const float* __restrict__ bias,
                     half_t* __restrict__ q, half_t* __restrict__ k,
                     half_t* __restrict__ v) {
  __shared__ half_t As[64][40];
  __shared__ half_t Bs[128][40];
  const int tid  = threadIdx.x;
  const int lane = tid & 31;
  const int w    = tid >> 5;
  const int wm   = w & 3;          // M sub-tile (16 rows each)
  const int wn   = w >> 2;         // N strip (64 cols each)
  const int row0 = blockIdx.y * 64;
  const int col0 = blockIdx.x * 128;

  v8f acc[4] = {};
  for (int k0 = 0; k0 < D_; k0 += 32) {
    __syncthreads();
#pragma unroll
    for (int t = 0; t < 8; ++t) {
      int i = tid + t * 256;                   // 64x32 A tile
      int m = i >> 5, kk = i & 31;
      As[m][kk] = (half_t)x[(size_t)(row0 + m) * D_ + k0 + kk];
    }
#pragma unroll
    for (int t = 0; t < 16; ++t) {
      int i = tid + t * 256;                   // 128x32 B tile (row-major over N)
      int n = i >> 5, kk = i & 31;
      Bs[n][kk] = (half_t)W[(size_t)(col0 + n) * D_ + k0 + kk];
    }
    __syncthreads();
    v16h af = load_a_frag<40>(&As[wm * 16][0], lane);
#pragma unroll
    for (int nb = 0; nb < 4; ++nb) {
      v16h bf = load_b_frag_nk<40>(&Bs[wn * 64 + nb * 16][0], lane);
      acc[nb] = wmma16(af, bf, acc[nb]);
    }
  }

  // Epilogue: each nb-tile sits inside one 64-col group -> h / which are
  // wave-uniform (no per-lane division).
  const int g     = (col0 >> 6) + wn;          // 64-column group index 0..23
  const int h     = g / 3;
  const int which = g % 3;
  half_t* dst = (which == 0) ? q : (which == 1) ? k : v;
#pragma unroll
  for (int nb = 0; nb < 4; ++nb) {
    const int dh  = nb * 16 + (lane & 15);
    const int col = col0 + wn * 64 + nb * 16 + (lane & 15);
    const float bv = bias[col];
#pragma unroll
    for (int r = 0; r < 8; ++r) {
      const int row = row0 + wm * 16 + r + 8 * (lane >> 4);
      const int bb  = row >> 11;               // row / S_
      const int s   = row & (S_ - 1);
      float val = acc[nb][r] + bv;
      if (which == 0) val *= 0.125f;           // 1/sqrt(64)
      dst[(((size_t)h * B_ + bb) * S_ + s) * DH_ + dh] = (half_t)val;
    }
  }
}

// ---------------------------------------------------------------------------
// Kernel 2: flash attention per (h,b). grid = (S/128, H*B), block = 256.
// Each wave owns a 16-query tile; block double-buffers 32-key K/V chunks in
// LDS via GLOBAL_LOAD_ASYNC_TO_LDS_B128 (ASYNCcnt software pipeline).
// ---------------------------------------------------------------------------
__global__ __launch_bounds__(256)
void attn_kernel(const half_t* __restrict__ q, const half_t* __restrict__ k,
                 const half_t* __restrict__ v, const float* __restrict__ shift,
                 const float* __restrict__ mask, half_t* __restrict__ o) {
  __shared__ half_t Ks[2][32][72];    // 2 x (32 keys x 64 dh), 144B rows
  __shared__ half_t Vs[2][32][72];
  __shared__ half_t Ps[8][16][40];    // per-wave P relayout scratch

  const int tid  = threadIdx.x;
  const int lane = tid & 31;
  const int w    = tid >> 5;
  const int h    = blockIdx.y >> 1;
  const int b    = blockIdx.y & 1;
  const int q0   = blockIdx.x * 128 + w * 16;
  const float hscale = exp2f(-(float)h);

  const size_t hb = (size_t)h * B_ + b;
  const half_t* qbase = q + (hb * S_ + (q0 + (lane & 15))) * DH_;
  const half_t* kbase = k + hb * S_ * DH_;
  const half_t* vbase = v + hb * S_ * DH_;

  // Q fragments (held for all key chunks). A-layout, K split per ISA table.
  v16h aq[2];
  {
    const int kb = (lane >> 4) * 8;
#pragma unroll
    for (int c = 0; c < 2; ++c) {
      const half_t* p = qbase + c * 32 + kb;
#pragma unroll
      for (int i = 0; i < 8; ++i)  aq[c][i] = p[i];
#pragma unroll
      for (int i = 8; i < 16; ++i) aq[c][i] = p[i + 8];
    }
  }

  // per-thread async global->LDS copy of one 16B vector of K and V
  const int ckey = tid >> 3;
  const int cd0  = (tid & 7) << 3;
  auto issue_chunk = [&](int kc, int buf) {
    unsigned ldsK = (unsigned)(size_t)(&Ks[buf][ckey][cd0]);
    unsigned ldsV = (unsigned)(size_t)(&Vs[buf][ckey][cd0]);
    uint64_t gk = (uint64_t)(size_t)(kbase + (size_t)(kc + ckey) * DH_ + cd0);
    uint64_t gv = (uint64_t)(size_t)(vbase + (size_t)(kc + ckey) * DH_ + cd0);
    asm volatile("global_load_async_to_lds_b128 %0, %1, off"
                 :: "v"(ldsK), "v"(gk) : "memory");
    asm volatile("global_load_async_to_lds_b128 %0, %1, off"
                 :: "v"(ldsV), "v"(gv) : "memory");
  };

  v8f   acc[4] = {};
  float m_i[8], l_i[8];
#pragma unroll
  for (int r = 0; r < 8; ++r) { m_i[r] = -1.0e30f; l_i[r] = 0.0f; }

  issue_chunk(0, 0);                       // prologue: chunk 0 -> buffer 0

  for (int kc = 0; kc < S_; kc += 32) {
    const int buf = (kc >> 5) & 1;
    // prefetch next chunk into the other buffer (safe: every thread already
    // passed the barrier ending the previous iteration's reads of buf^1)
    const int kcn = (kc + 32 < S_) ? (kc + 32) : 0;   // last issue unused
    issue_chunk(kcn, buf ^ 1);
    // wait for the oldest two async ops (current chunk's K and V; in-order)
    asm volatile("s_wait_asynccnt 0x2" ::: "memory");
    __syncthreads();

    // logits tiles: 16 queries x 2x16 keys, contract over dh=64 (2 WMMAs each)
    v8f lg[2];
#pragma unroll
    for (int t = 0; t < 2; ++t) {
      lg[t] = (v8f){};
#pragma unroll
      for (int c = 0; c < 2; ++c) {
        // B[K=d][N=key] = Ks[key][d]  -> Mem[N][K] form
        v16h bk = load_b_frag_nk<72>(&Ks[buf][t * 16][c * 32], lane);
        lg[t] = wmma16(aq[c], bk, lg[t]);
      }
    }

    // fused shift + mask (C-layout: N=lane&15, M=r+8*(lane>=16))
#pragma unroll
    for (int t = 0; t < 2; ++t) {
      const int key = kc + t * 16 + (lane & 15);
#pragma unroll
      for (int r = 0; r < 8; ++r) {
        const int query = q0 + r + 8 * (lane >> 4);
        const size_t off = ((size_t)b * S_ + query) * S_ + key;
        const float sv = shift[off];
        const float mv = mask[off];
        if (r == 0 && kc + 32 < S_) {
          __builtin_prefetch((const void*)(shift + off + 32), 0, 0);
          __builtin_prefetch((const void*)(mask + off + 32), 0, 0);
        }
        float xv = lg[t][r] - hscale * sv;
        xv = xv * mv - 1.0e8f * (1.0f - mv);
        lg[t][r] = xv;
      }
    }

    // online softmax, rows reduced across the 16 lanes of each half-wave
#pragma unroll
    for (int r = 0; r < 8; ++r) {
      float cm = fmaxf(lg[0][r], lg[1][r]);
#pragma unroll
      for (int m = 8; m >= 1; m >>= 1) cm = fmaxf(cm, __shfl_xor(cm, m, 32));
      const float nm   = fmaxf(m_i[r], cm);
      const float corr = __expf(m_i[r] - nm);
      const float p0 = __expf(lg[0][r] - nm);
      const float p1 = __expf(lg[1][r] - nm);
      lg[0][r] = p0; lg[1][r] = p1;
      float rs = p0 + p1;
#pragma unroll
      for (int m = 8; m >= 1; m >>= 1) rs += __shfl_xor(rs, m, 32);
      l_i[r] = l_i[r] * corr + rs;
      m_i[r] = nm;
#pragma unroll
      for (int n = 0; n < 4; ++n) acc[n][r] *= corr;
    }

    // relayout P (C-layout -> A-layout) through per-wave LDS scratch
#pragma unroll
    for (int t = 0; t < 2; ++t)
#pragma unroll
      for (int r = 0; r < 8; ++r)
        Ps[w][r + 8 * (lane >> 4)][t * 16 + (lane & 15)] = (half_t)lg[t][r];
    // same-wave LDS ops are in-order; compiler inserts s_wait_dscnt for reuse

    v16h ap = load_a_frag<40>(&Ps[w][0][0], lane);
#pragma unroll
    for (int n = 0; n < 4; ++n) {
      // B[K=key][N=dh] = Vs[key][dh]  -> Mem[K][N] form
      v16h bv = load_b_frag_kn<72>(&Vs[buf][0][n * 16], lane);
      acc[n] = wmma16(ap, bv, acc[n]);
    }
    __syncthreads();   // protect buf before next iteration's async overwrite
  }

  // normalize and store o[b][s][h*64+dh] as f16
#pragma unroll
  for (int n = 0; n < 4; ++n) {
    const int dcol = n * 16 + (lane & 15);
#pragma unroll
    for (int r = 0; r < 8; ++r) {
      const int query = q0 + r + 8 * (lane >> 4);
      const float val = acc[n][r] / l_i[r];
      o[((size_t)b * S_ + query) * D_ + h * DH_ + dcol] = (half_t)val;
    }
  }
}

// ---------------------------------------------------------------------------
// Kernel 3: out = o @ Wo^T + bo (f32 out). grid = (512/128, 4096/64), 256 thr.
// ---------------------------------------------------------------------------
__global__ __launch_bounds__(256)
void out_proj_kernel(const half_t* __restrict__ o, const float* __restrict__ Wo,
                     const float* __restrict__ bo, float* __restrict__ out) {
  __shared__ half_t As[64][40];
  __shared__ half_t Bs[128][40];
  const int tid  = threadIdx.x;
  const int lane = tid & 31;
  const int w    = tid >> 5;
  const int wm   = w & 3;
  const int wn   = w >> 2;
  const int row0 = blockIdx.y * 64;
  const int col0 = blockIdx.x * 128;

  v8f acc[4] = {};
  for (int k0 = 0; k0 < D_; k0 += 32) {
    __syncthreads();
    {  // A tile: 64x32 halves, one uint4 (8 halves) per thread
      const int m  = tid >> 2;
      const int kl = (tid & 3) << 3;
      *(uint4*)(&As[m][kl]) =
          *(const uint4*)(o + (size_t)(row0 + m) * D_ + k0 + kl);
    }
#pragma unroll
    for (int t = 0; t < 16; ++t) {
      int i = tid + t * 256;
      int n = i >> 5, kk = i & 31;
      Bs[n][kk] = (half_t)Wo[(size_t)(col0 + n) * D_ + k0 + kk];
    }
    __syncthreads();
    v16h af = load_a_frag<40>(&As[wm * 16][0], lane);
#pragma unroll
    for (int nb = 0; nb < 4; ++nb) {
      v16h bf = load_b_frag_nk<40>(&Bs[wn * 64 + nb * 16][0], lane);
      acc[nb] = wmma16(af, bf, acc[nb]);
    }
  }
#pragma unroll
  for (int nb = 0; nb < 4; ++nb) {
    const int col = col0 + wn * 64 + nb * 16 + (lane & 15);
    const float bv = bo[col];
#pragma unroll
    for (int r = 0; r < 8; ++r) {
      const int row = row0 + wm * 16 + r + 8 * (lane >> 4);
      out[(size_t)row * D_ + col] = acc[nb][r] + bv;
    }
  }
}

// ---------------------------------------------------------------------------
extern "C" void kernel_launch(void* const* d_in, const int* in_sizes, int n_in,
                              void* d_out, int out_size, void* d_ws,
                              size_t ws_size, hipStream_t stream) {
  const float* x     = (const float*)d_in[0];
  const float* shift = (const float*)d_in[1];
  const float* mask  = (const float*)d_in[2];
  const float* W     = (const float*)d_in[3];
  const float* bias  = (const float*)d_in[4];
  const float* Wo    = (const float*)d_in[5];
  const float* bo    = (const float*)d_in[6];
  float* out = (float*)d_out;

  // workspace: q,k,v [H][B][S][DH] f16 (4 MB each) + o [B][S][D] f16 (4 MB)
  const size_t per = (size_t)H_ * B_ * S_ * DH_;   // 2,097,152 halves
  half_t* qws = (half_t*)d_ws;
  half_t* kws = qws + per;
  half_t* vws = kws + per;
  half_t* ows = vws + per;

  qkv_gemm_kernel<<<dim3(N3_ / 128, (B_ * S_) / 64), 256, 0, stream>>>(
      x, W, bias, qws, kws, vws);
  attn_kernel<<<dim3(S_ / 128, H_ * B_), 256, 0, stream>>>(
      qws, kws, vws, shift, mask, ows);
  out_proj_kernel<<<dim3(D_ / 128, (B_ * S_) / 64), 256, 0, stream>>>(
      ows, Wo, bo, out);
}
